// SingleAttention_53317724012823
// MI455X (gfx1250) — compile-verified, matrix operand order checked by reading, not run on hardware
//
#include <hip/hip_runtime.h>
#include <hip/hip_bf16.h>

// ---------------------------------------------------------------------------
// SingleAttention for MI455X (gfx1250, wave32, WMMA f16 -> f32 accumulate)
//   c   = embd @ W^T + b                  [8, 2048, 1024]
//   S   = softmax(c @ c^T / 32)           [8, 2048, 2048]   (never hits HBM)
//   out = S @ c                           [8, 2048, 1024]
// ---------------------------------------------------------------------------

#define BB    8
#define NN    2048
#define EE    1024
#define OO    1024
#define SCALE (1.0f / 32.0f)   // 1/sqrt(1024)

typedef __attribute__((ext_vector_type(16))) _Float16 v16h;
typedef __attribute__((ext_vector_type(8)))  _Float16 v8h;
typedef __attribute__((ext_vector_type(8)))  float    v8f;

typedef __attribute__((ext_vector_type(4))) unsigned int u32x4;
typedef __attribute__((ext_vector_type(8))) int          i32x8;
typedef __attribute__((ext_vector_type(4))) int          i32x4;

#if __has_builtin(__builtin_amdgcn_tensor_load_to_lds) && \
    __has_builtin(__builtin_amdgcn_s_wait_tensorcnt)
#define SATTN_HAVE_TDM 1
#else
#define SATTN_HAVE_TDM 0
#endif

static __device__ __forceinline__ v16h mk16(v8h lo, v8h hi) {
    v16h r;
#pragma unroll
    for (int i = 0; i < 8; ++i) { r[i] = lo[i]; r[i + 8] = hi[i]; }
    return r;
}

static __device__ __forceinline__ v8f wmma_f16(v16h a, v16h b, v8f c) {
    return __builtin_amdgcn_wmma_f32_16x16x32_f16(false, a, false, b,
                                                  (short)0, c, false, false);
}

// ===========================================================================
// Kernel 1: c = embd @ W^T + bias, stored as f16 row-major and f16 transposed.
//   grid = (BN/128, OO/64), block = 256 (8 waves), per-wave tile 32x32.
// ===========================================================================
__global__ void __launch_bounds__(256)
sattn_linear_kernel(const float* __restrict__ embd,   // [B*N, E] f32
                    const float* __restrict__ Wt,     // [O, E]   f32
                    const float* __restrict__ bias,   // [O]      f32
                    _Float16* __restrict__ cOut,      // [B*N, O] f16
                    _Float16* __restrict__ cTOut)     // [B][O][N] f16
{
    __shared__ _Float16 Asm[128][32];  // embd tile (f16)   8 KB
    __shared__ _Float16 Bsm[64][32];   // W    tile (f16)   4 KB

    const int t      = threadIdx.x;
    const int w      = t >> 5;          // wave 0..7
    const int lane   = t & 31;
    const int rlane  = lane & 15;
    const int rhalf  = lane >> 4;       // 0/1 -> K sub-halves per WMMA layout
    const int wave_m = w & 3;           // 4 waves over M (32 rows each)
    const int wave_n = w >> 2;          // 2 waves over N (32 cols each)

    const int tileM = blockIdx.x * 128; // row block in [0, B*N)
    const int tileN = blockIdx.y * 64;  // col block in [0, O)

    v8f acc[2][2] = {};

    // staging roles
    const int arow = t >> 1, ahalf = t & 1;   // A: 128 rows x 2 halves of 16
    const int brow = t >> 2, bq    = t & 3;   // B: 64 rows  x 4 chunks of 8

    const float* srcA = embd + (size_t)(tileM + arow) * EE + ahalf * 16;
    const float* srcB = Wt   + (size_t)(tileN + brow) * EE + bq * 8;

    for (int kk = 0; kk < EE; kk += 32) {
        // ---- stage A tile: 16 f32 -> 16 f16 per thread ----
        {
            const float4* p = (const float4*)(srcA + kk);
            float4 f0 = p[0], f1 = p[1], f2 = p[2], f3 = p[3];
            __builtin_prefetch(srcA + kk + 32, 0, 0);  // global_prefetch_b8
            v8h h0, h1;
            h0[0]=(_Float16)f0.x; h0[1]=(_Float16)f0.y; h0[2]=(_Float16)f0.z; h0[3]=(_Float16)f0.w;
            h0[4]=(_Float16)f1.x; h0[5]=(_Float16)f1.y; h0[6]=(_Float16)f1.z; h0[7]=(_Float16)f1.w;
            h1[0]=(_Float16)f2.x; h1[1]=(_Float16)f2.y; h1[2]=(_Float16)f2.z; h1[3]=(_Float16)f2.w;
            h1[4]=(_Float16)f3.x; h1[5]=(_Float16)f3.y; h1[6]=(_Float16)f3.z; h1[7]=(_Float16)f3.w;
            *(v8h*)&Asm[arow][ahalf * 16]     = h0;
            *(v8h*)&Asm[arow][ahalf * 16 + 8] = h1;
        }
        // ---- stage B tile: 8 f32 -> 8 f16 per thread ----
        {
            const float4* p = (const float4*)(srcB + kk);
            float4 f0 = p[0], f1 = p[1];
            __builtin_prefetch(srcB + kk + 32, 0, 0);
            v8h h;
            h[0]=(_Float16)f0.x; h[1]=(_Float16)f0.y; h[2]=(_Float16)f0.z; h[3]=(_Float16)f0.w;
            h[4]=(_Float16)f1.x; h[5]=(_Float16)f1.y; h[6]=(_Float16)f1.z; h[7]=(_Float16)f1.w;
            *(v8h*)&Bsm[brow][bq * 8] = h;
        }
        __syncthreads();

        // ---- fragments + WMMA (A/B per-lane: two 8-half runs, per ISA layout) ----
        v16h afrag[2], bfrag[2];
#pragma unroll
        for (int mt = 0; mt < 2; ++mt) {
            const int r = wave_m * 32 + mt * 16 + rlane;
            afrag[mt] = mk16(*(const v8h*)&Asm[r][rhalf * 8],
                             *(const v8h*)&Asm[r][16 + rhalf * 8]);
        }
#pragma unroll
        for (int nt = 0; nt < 2; ++nt) {
            const int cN = wave_n * 32 + nt * 16 + rlane;
            bfrag[nt] = mk16(*(const v8h*)&Bsm[cN][rhalf * 8],
                             *(const v8h*)&Bsm[cN][16 + rhalf * 8]);
        }
#pragma unroll
        for (int mt = 0; mt < 2; ++mt)
#pragma unroll
            for (int nt = 0; nt < 2; ++nt)
                acc[mt][nt] = wmma_f16(afrag[mt], bfrag[nt], acc[mt][nt]);
        __syncthreads();
    }

    // ---- epilogue: +bias, cvt f16, store c (scalar, lane-coalesced) and
    //      cT (the 8 accumulator rows of a lane are contiguous n -> one b128) ----
#pragma unroll
    for (int nt = 0; nt < 2; ++nt) {
        const int gn = tileN + wave_n * 32 + nt * 16 + rlane;
        const float bv = bias[gn];
#pragma unroll
        for (int mt = 0; mt < 2; ++mt) {
            v8h hv;
#pragma unroll
            for (int r = 0; r < 8; ++r)
                hv[r] = (_Float16)(acc[mt][nt][r] + bv);
            const int gmBase = tileM + wave_m * 32 + mt * 16 + 8 * rhalf; // 8-aligned
#pragma unroll
            for (int r = 0; r < 8; ++r)
                cOut[(size_t)(gmBase + r) * OO + gn] = hv[r];
            const int batch = gmBase >> 11;          // /N (tile stays in one batch)
            const int rb    = gmBase & (NN - 1);
            *(v8h*)&cTOut[((size_t)batch * OO + gn) * NN + rb] = hv;
        }
    }
}

// ===========================================================================
// Kernel 2: fused  S = softmax(c c^T / 32),  out = S @ c
//   grid = B * N/16 blocks, 256 threads (8 waves).
//   Q rows pulled into LDS by the Tensor Data Mover (16x1024 f16 2D tile).
//   Phase 1: wave w -> key cols [w*256, w*256+256), S strip in registers.
//   Softmax: register + shfl + LDS cross-wave reduction; P (f16) -> LDS.
//   Phase 2: wave w -> out cols [w*128, w*128+128), B from cT (contiguous).
// ===========================================================================
__global__ void __launch_bounds__(256)
sattn_attn_kernel(const _Float16* __restrict__ cbuf,   // [B*N, O]
                  const _Float16* __restrict__ ctbuf,  // [B][O][N]
                  float* __restrict__ out)             // [B*N, O] f32
{
    extern __shared__ _Float16 smem[];
    _Float16* Pbuf = smem;                 // [16][2048] un-normalized probs (64 KB)
    _Float16* Qbuf = smem + 16 * 2048;     // [16][1024] query rows of c     (32 KB)
    __shared__ float redbuf[16][8];        // cross-wave max/sum scratch

    const int t     = threadIdx.x;
    const int w     = t >> 5;
    const int lane  = t & 31;
    const int rlane = lane & 15;
    const int rhalf = lane >> 4;

    const int batch = blockIdx.x >> 7;         // / (N/16)
    const int q0    = (blockIdx.x & 127) * 16; // first query row in batch
    const size_t rowbase = (size_t)(batch * NN + q0);

    // ---- load the 16 query rows of c into LDS ----
#if SATTN_HAVE_TDM
    if (w == 0) {
        // Tensor DMA descriptor (D#): 2D tile, 16 rows x 1024 f16, row stride O.
        const unsigned long long ga =
            (unsigned long long)(const void*)(cbuf + rowbase * OO);
        u32x4 g0;
        g0[0] = 1u;                                        // count=1 (valid D#)
        g0[1] = (unsigned int)(size_t)Qbuf;                // lds_addr (bytes)
        g0[2] = (unsigned int)(ga & 0xFFFFFFFFu);          // global_addr[31:0]
        g0[3] = (unsigned int)((ga >> 32) & 0x1FFFFFFu)    // global_addr[56:32]
                | (2u << 30);                              // type = 2 ("image")
        i32x8 g1;
        g1[0] = 0x00010000;          // workgroup_mask=0, data_size=1 (2 bytes)
        g1[1] = (int)(1024u << 16);  // tensor_dim0 = 1024  (bits 79:48, low part)
        g1[2] = (int)(2048u << 16);  // dim0 hi = 0 | tensor_dim1 = 2048 (low part)
        g1[3] = (int)(1024u << 16);  // dim1 hi = 0 | tile_dim0 = 1024
        g1[4] = 16;                  // tile_dim1 = 16, tile_dim2 = 0
        g1[5] = 1024;                // tensor_dim0_stride = 1024 elements
        g1[6] = 0;                   // stride hi | tensor_dim1_stride lo = 0
        g1[7] = 0;
        i32x4 g2 = {0, 0, 0, 0};     // 2D tensor: groups 2/3 unused
        i32x4 g3 = {0, 0, 0, 0};
#if defined(__clang_major__) && (__clang_major__ >= 23)
        i32x8 g4 = {0, 0, 0, 0, 0, 0, 0, 0};
        __builtin_amdgcn_tensor_load_to_lds(g0, g1, g2, g3, g4, 0);
#else
        __builtin_amdgcn_tensor_load_to_lds(g0, g1, g2, g3, 0);
#endif
        __builtin_amdgcn_s_wait_tensorcnt(0);
    }
#else
    {
        const int r = t >> 4, seg = t & 15;
        const _Float16* src = cbuf + (rowbase + r) * OO + seg * 64;
        _Float16* dst = Qbuf + r * 1024 + seg * 64;
#pragma unroll
        for (int i = 0; i < 8; ++i)
            *(v8h*)(dst + i * 8) = *(const v8h*)(src + i * 8);
    }
#endif
    __syncthreads();

    // ---------------- Phase 1: scores for 256 key cols per wave ----------------
    v8f sacc[16] = {};
    {
        const size_t keybase = (size_t)(batch * NN + w * 256);
        for (int kk = 0; kk < OO; kk += 32) {
            const v16h a = mk16(*(const v8h*)&Qbuf[rlane * 1024 + kk + rhalf * 8],
                                *(const v8h*)&Qbuf[rlane * 1024 + kk + 16 + rhalf * 8]);
#pragma unroll
            for (int nt = 0; nt < 16; ++nt) {
                const _Float16* kr = cbuf + (keybase + nt * 16 + rlane) * OO + kk;
                const v16h b = mk16(*(const v8h*)(kr + rhalf * 8),
                                    *(const v8h*)(kr + 16 + rhalf * 8));
                sacc[nt] = wmma_f16(a, b, sacc[nt]);
            }
        }
    }

    // ---------------- softmax (rows r + 8*rhalf live on this lane) ----------------
    float gmax[8], gsum[8];
#pragma unroll
    for (int r = 0; r < 8; ++r) {
        float mx = -1e30f;
#pragma unroll
        for (int nt = 0; nt < 16; ++nt) mx = fmaxf(mx, sacc[nt][r]);
#pragma unroll
        for (int m = 1; m < 16; m <<= 1) mx = fmaxf(mx, __shfl_xor(mx, m, 16));
        gmax[r] = mx;
    }
    if (rlane == 0)
#pragma unroll
        for (int r = 0; r < 8; ++r) redbuf[r + 8 * rhalf][w] = gmax[r];
    __syncthreads();
#pragma unroll
    for (int r = 0; r < 8; ++r) {
        float mx = -1e30f;
#pragma unroll
        for (int ww = 0; ww < 8; ++ww) mx = fmaxf(mx, redbuf[r + 8 * rhalf][ww]);
        gmax[r] = mx;
    }
    __syncthreads();  // redbuf reused for sums

#pragma unroll
    for (int r = 0; r < 8; ++r) {
        float s = 0.0f;
        const int prow = (r + 8 * rhalf) * 2048 + w * 256 + rlane;
#pragma unroll
        for (int nt = 0; nt < 16; ++nt) {
            const float e = __expf((sacc[nt][r] - gmax[r]) * SCALE);
            s += e;
            Pbuf[prow + nt * 16] = (_Float16)e;
        }
#pragma unroll
        for (int m = 1; m < 16; m <<= 1) s += __shfl_xor(s, m, 16);
        gsum[r] = s;
    }
    if (rlane == 0)
#pragma unroll
        for (int r = 0; r < 8; ++r) redbuf[r + 8 * rhalf][w] = gsum[r];
    __syncthreads();
#pragma unroll
    for (int r = 0; r < 8; ++r) {
        float s = 0.0f;
#pragma unroll
        for (int ww = 0; ww < 8; ++ww) s += redbuf[r + 8 * rhalf][ww];
        gsum[r] = s;
    }
    __syncthreads();  // Pbuf fully written before phase 2 reads it

    // ---------------- Phase 2: out = P @ c  (B from cT, contiguous rows) ----------------
    v8f oacc[8] = {};
    {
        const size_t ctb = (size_t)batch * OO;
        for (int kk = 0; kk < NN; kk += 32) {
            const v16h a = mk16(*(const v8h*)&Pbuf[rlane * 2048 + kk + rhalf * 8],
                                *(const v8h*)&Pbuf[rlane * 2048 + kk + 16 + rhalf * 8]);
#pragma unroll
            for (int nt = 0; nt < 8; ++nt) {
                const int ocol = w * 128 + nt * 16 + rlane;
                const _Float16* br = ctbuf + (ctb + ocol) * NN + kk;
                const v16h b = mk16(*(const v8h*)(br + rhalf * 8),
                                    *(const v8h*)(br + 16 + rhalf * 8));
                oacc[nt] = wmma_f16(a, b, oacc[nt]);
            }
        }
    }

    // ---- epilogue: divide by row sum, store f32 ----
#pragma unroll
    for (int nt = 0; nt < 8; ++nt) {
        const int ocol = w * 128 + nt * 16 + rlane;
#pragma unroll
        for (int r = 0; r < 8; ++r) {
            const int m = r + 8 * rhalf;
            out[(rowbase + m) * OO + ocol] = oacc[nt][r] / gsum[r];
        }
    }
}

// ===========================================================================
extern "C" void kernel_launch(void* const* d_in, const int* in_sizes, int n_in,
                              void* d_out, int out_size, void* d_ws, size_t ws_size,
                              hipStream_t stream) {
    (void)in_sizes; (void)n_in; (void)out_size; (void)ws_size;
    const float* embd = (const float*)d_in[0];
    const float* Wt   = (const float*)d_in[1];
    const float* bias = (const float*)d_in[2];
    float* out = (float*)d_out;

    _Float16* cbuf  = (_Float16*)d_ws;                    // 32 MB  [B*N, O]
    _Float16* ctbuf = cbuf + (size_t)BB * NN * OO;        // 32 MB  [B][O][N]

    dim3 g1((BB * NN) / 128, OO / 64);
    sattn_linear_kernel<<<g1, 256, 0, stream>>>(embd, Wt, bias, cbuf, ctbuf);

    const size_t shmem = (size_t)(16 * 2048 + 16 * 1024) * sizeof(_Float16); // 96 KB
    sattn_attn_kernel<<<dim3(BB * (NN / 16)), 256, shmem, stream>>>(cbuf, ctbuf, out);
}
